// UGCASeqModel_11175504904234
// MI455X (gfx1250) — compile-verified, matrix operand order checked by reading, not run on hardware
//
#include <hip/hip_runtime.h>
#include <hip/hip_bf16.h>
#include <stdint.h>

// ---------------------------------------------------------------------------
// UGCA seq model forward, CDNA5 (gfx1250) — bf16 WMMA everywhere that matters.
// Workspace requirement: ~420 MB.
// ---------------------------------------------------------------------------

#define DM   512
#define DK   128
#define NH   4
#define RANKM 10
#define MUT  512

typedef __attribute__((ext_vector_type(16))) __bf16    bf16x16;
typedef __attribute__((ext_vector_type(8)))  float     v8f;
typedef __attribute__((ext_vector_type(8)))  unsigned  v8u;

__device__ __forceinline__ uint16_t f2bf(float f) {
  uint32_t u = __builtin_bit_cast(uint32_t, f);
  uint32_t r = u + 0x7FFFu + ((u >> 16) & 1u);   // round-to-nearest-even
  return (uint16_t)(r >> 16);
}

// Documented 16-bit A/B fragment K mapping (ISA 7.12.2):
// lanes 0-15: VGPR j -> K = {0,2,4,6,16,18,20,22}[j] ; lanes 16-31: +8
__device__ __forceinline__ int kmap(int j, int half) {
  return ((j < 4) ? (2 * j) : (16 + 2 * (j - 4))) + (half ? 8 : 0);
}

__device__ __forceinline__ v8f wmma_bf(v8u a, v8u b, v8f c) {
  return __builtin_amdgcn_wmma_f32_16x16x32_bf16(
      false, __builtin_bit_cast(bf16x16, a),
      false, __builtin_bit_cast(bf16x16, b),
      (short)0, c, false, false);
}

// ---------------------------------------------------------------------------
// Generic GEMM: C(MxN) = act(A(MxK) @ W(KxN) + bias)
//   fp32 A/W converted to bf16 in LDS, fp32 accumulate via v_wmma.
//   ACT: 0=none 1=silu.  OUTMODE: 0=f32, 1=bf16, 2=bf16 transposed per-head
//   (layout (B,H,dk,T), used for V so attention B-fragments are contiguous).
// ---------------------------------------------------------------------------
template <int ACT, int OUTMODE>
__global__ __launch_bounds__(128)
void gemm_wmma(const float* __restrict__ A, const float* __restrict__ W,
               const float* __restrict__ bias, void* __restrict__ Cout,
               int M, int N, int K, int T) {
  __shared__ uint16_t As[64][36];   // [m][k], pitch 36 keeps dword align
  __shared__ uint16_t Bs[64][36];   // [n][k] (transposed weight tile)

  const int tid  = threadIdx.x;
  const int lane = tid & 31, wave = tid >> 5;
  const int half = lane >> 4, l16 = lane & 15;
  const int m0 = blockIdx.y * 64, n0 = blockIdx.x * 64;
  const int wm = (wave >> 1) * 32, wn = (wave & 1) * 32;

  const v8f zf = {0.f,0.f,0.f,0.f,0.f,0.f,0.f,0.f};
  v8f acc[2][2];
  acc[0][0] = zf; acc[0][1] = zf; acc[1][0] = zf; acc[1][1] = zf;

  const int ar = tid >> 3;          // 0..15
  const int ac = (tid & 7) << 2;    // 0..28
  const int wr = tid >> 4;          // 0..7
  const int wc = (tid & 15) << 2;   // 0..60

  for (int k0 = 0; k0 < K; k0 += 32) {
    __syncthreads();
    // ---- stage A tile (64 x 32), fp32 -> bf16
#pragma unroll
    for (int rr = ar; rr < 64; rr += 16) {
      const int grow = m0 + rr;
      float f0 = 0.f, f1 = 0.f, f2 = 0.f, f3 = 0.f;
      if (grow < M) {
        const float* ap = A + (size_t)grow * K + k0 + ac;
        if (k0 + ac + 3 < K) {
          float4 v = *(const float4*)ap;
          f0 = v.x; f1 = v.y; f2 = v.z; f3 = v.w;
        } else {
          if (k0 + ac + 0 < K) f0 = ap[0];
          if (k0 + ac + 1 < K) f1 = ap[1];
          if (k0 + ac + 2 < K) f2 = ap[2];
          if (k0 + ac + 3 < K) f3 = ap[3];
        }
        if (k0 + 32 < K) __builtin_prefetch(ap + 32, 0, 0);  // global_prefetch_b8
      }
      As[rr][ac + 0] = f2bf(f0); As[rr][ac + 1] = f2bf(f1);
      As[rr][ac + 2] = f2bf(f2); As[rr][ac + 3] = f2bf(f3);
    }
    // ---- stage W tile (32 x 64) transposed into Bs[n][k]
#pragma unroll
    for (int kk = wr; kk < 32; kk += 8) {
      const int gk = k0 + kk;
      float f0 = 0.f, f1 = 0.f, f2 = 0.f, f3 = 0.f;
      if (gk < K) {
        const float* wp = W + (size_t)gk * N + n0 + wc;
        if (n0 + wc + 3 < N) {
          float4 v = *(const float4*)wp;
          f0 = v.x; f1 = v.y; f2 = v.z; f3 = v.w;
        } else {
          if (n0 + wc + 0 < N) f0 = wp[0];
          if (n0 + wc + 1 < N) f1 = wp[1];
          if (n0 + wc + 2 < N) f2 = wp[2];
          if (n0 + wc + 3 < N) f3 = wp[3];
        }
      }
      Bs[wc + 0][kk] = f2bf(f0); Bs[wc + 1][kk] = f2bf(f1);
      Bs[wc + 2][kk] = f2bf(f2); Bs[wc + 3][kk] = f2bf(f3);
    }
    __syncthreads();

    // ---- fragments + 4 WMMA per K-step per wave
    v8u fa[2], fb[2];
#pragma unroll
    for (int t = 0; t < 2; t++) {
      const uint32_t* ap = (const uint32_t*)(&As[wm + t * 16 + l16][0]);
      const uint32_t* bp = (const uint32_t*)(&Bs[wn + t * 16 + l16][0]);
#pragma unroll
      for (int j = 0; j < 8; j++) {
        const int kb = kmap(j, half) >> 1;
        fa[t][j] = ap[kb];
        fb[t][j] = bp[kb];
      }
    }
#pragma unroll
    for (int i = 0; i < 2; i++)
#pragma unroll
      for (int j = 0; j < 2; j++)
        acc[i][j] = wmma_bf(fa[i], fb[j], acc[i][j]);
  }

  // ---- epilogue: C layout VGPR e -> row e + half*8 ; lane&15 -> col
#pragma unroll
  for (int i = 0; i < 2; i++) {
#pragma unroll
    for (int j = 0; j < 2; j++) {
      const int col = n0 + wn + j * 16 + l16;
      const float bs = (bias != nullptr && col < N) ? bias[col] : 0.f;
#pragma unroll
      for (int e = 0; e < 8; e++) {
        const int row = m0 + wm + i * 16 + e + half * 8;
        float v = acc[i][j][e] + bs;
        if (ACT == 1) v = v * (1.f / (1.f + __expf(-v)));   // silu
        if (row < M && col < N) {
          if (OUTMODE == 0) {
            ((float*)Cout)[(size_t)row * N + col] = v;
          } else if (OUTMODE == 1) {
            ((uint16_t*)Cout)[(size_t)row * N + col] = f2bf(v);
          } else {  // per-head transposed (B,H,dk,T)
            const int b_ = row / T, t_ = row % T;
            const int h_ = col / DK, kd = col % DK;
            ((uint16_t*)Cout)[(((size_t)b_ * NH + h_) * DK + kd) * T + t_] = f2bf(v);
          }
        }
      }
    }
  }
}

// ---------------------------------------------------------------------------
// Flash-style gated cross attention, one 16-query tile per wave.
//   O[b,q,h*dk+..] = softmax(scale*QK^T + lgk[key]) @ V
// (query-side gate bias is row-constant -> cancels in softmax; masks all-true)
// Q,K: bf16 (B,T,DM); Vt: bf16 (B,H,DK,Tk); O: f32 (B,Tq,DM)
// ---------------------------------------------------------------------------
__global__ __launch_bounds__(128)
void attn_wmma(const uint16_t* __restrict__ Q, const uint16_t* __restrict__ Km,
               const uint16_t* __restrict__ Vt, const float* __restrict__ lgk,
               float* __restrict__ O, int Bc, int Tq, int Tk, float scale) {
  __shared__ uint16_t Ps[4][16][36];  // per-wave prob redistribution tile

  const int tid = threadIdx.x, lane = tid & 31, wave = tid >> 5;
  const int half = lane >> 4, l16 = lane & 15;
  const int qtiles = Tq >> 4;
  int gw = blockIdx.x * 4 + wave;
  if (gw >= Bc * NH * qtiles) return;
  const int qt = gw % qtiles; gw /= qtiles;
  const int h = gw % NH; const int b = gw / NH;

  // preload Q A-fragments (16x128 in 4 chunks of K=32)
  const uint16_t* qb = Q + ((size_t)b * Tq + qt * 16 + l16) * DM + h * DK;
  v8u qf[4];
#pragma unroll
  for (int c = 0; c < 4; c++)
#pragma unroll
    for (int j = 0; j < 8; j++)
      qf[c][j] = *(const uint32_t*)(qb + c * 32 + kmap(j, half));

  const v8f zf = {0.f,0.f,0.f,0.f,0.f,0.f,0.f,0.f};
  float mrow[8], srow[8];
  v8f oacc[8];
#pragma unroll
  for (int e = 0; e < 8; e++) { mrow[e] = -3.0e38f; srow[e] = 0.f; }
#pragma unroll
  for (int j = 0; j < 8; j++) oacc[j] = zf;

  uint16_t* psw = &Ps[wave][0][0];

  for (int kc = 0; kc < Tk; kc += 32) {
    // ---- scores: S (16 x 32) = Q @ K^T  (two 16-key WMMA columns)
    v8f s0 = zf, s1 = zf;
#pragma unroll
    for (int c = 0; c < 4; c++) {
      const uint16_t* kp = Km + ((size_t)b * Tk + kc + l16) * DM + h * DK + c * 32;
      v8u f0, f1;
#pragma unroll
      for (int j = 0; j < 8; j++) {
        const int kb = kmap(j, half);
        f0[j] = *(const uint32_t*)(kp + kb);
        f1[j] = *(const uint32_t*)(kp + 16 * DM + kb);
      }
      s0 = wmma_bf(qf[c], f0, s0);
      s1 = wmma_bf(qf[c], f1, s1);
    }
    const float g0 = lgk[(size_t)b * Tk + kc + l16];
    const float g1 = lgk[(size_t)b * Tk + kc + 16 + l16];

    // ---- online softmax (rows live in this lane's half-group)
    float al[8];
#pragma unroll
    for (int e = 0; e < 8; e++) {
      float a = s0[e] * scale + g0;
      float bsc = s1[e] * scale + g1;
      float cm = fmaxf(a, bsc);
      cm = fmaxf(cm, __shfl_xor(cm, 1, 32));
      cm = fmaxf(cm, __shfl_xor(cm, 2, 32));
      cm = fmaxf(cm, __shfl_xor(cm, 4, 32));
      cm = fmaxf(cm, __shfl_xor(cm, 8, 32));
      const float mn = fmaxf(mrow[e], cm);
      al[e] = __expf(mrow[e] - mn);
      mrow[e] = mn;
      const float p0 = __expf(a - mn), p1 = __expf(bsc - mn);
      float rs = p0 + p1;
      rs += __shfl_xor(rs, 1, 32);
      rs += __shfl_xor(rs, 2, 32);
      rs += __shfl_xor(rs, 4, 32);
      rs += __shfl_xor(rs, 8, 32);
      srow[e] = srow[e] * al[e] + rs;
      // C-layout -> LDS (to be re-read in A-layout for the PV WMMA)
      psw[(e + half * 8) * 36 + l16]      = f2bf(p0);
      psw[(e + half * 8) * 36 + 16 + l16] = f2bf(p1);
    }
#pragma unroll
    for (int j = 0; j < 8; j++)
#pragma unroll
      for (int e = 0; e < 8; e++) oacc[j][e] *= al[e];

    // ---- probs A-fragment
    v8u pf;
    const uint32_t* pp = (const uint32_t*)(psw + l16 * 36);
#pragma unroll
    for (int j = 0; j < 8; j++) pf[j] = pp[kmap(j, half) >> 1];

    // ---- PV: O(16 x 128) += P(16x32) @ V(32x128) (8 N-tiles)
#pragma unroll
    for (int j = 0; j < 8; j++) {
      const uint16_t* vp =
          Vt + (((size_t)b * NH + h) * DK + j * 16 + l16) * Tk + kc;
      v8u vf;
#pragma unroll
      for (int q = 0; q < 8; q++) vf[q] = *(const uint32_t*)(vp + kmap(q, half));
      oacc[j] = wmma_bf(pf, vf, oacc[j]);
    }
  }

  // ---- normalize + write
#pragma unroll
  for (int j = 0; j < 8; j++)
#pragma unroll
    for (int e = 0; e < 8; e++) {
      const int qrow = qt * 16 + e + half * 8;
      O[((size_t)b * Tq + qrow) * DM + h * DK + j * 16 + l16] =
          oacc[j][e] / srow[e];
    }
}

// ---------------------------------------------------------------------------
// Evidential gate: lg[tok] = log(max(sigmoid(mu)*exp(-2*beta/(v*(a-1))),1e-6))
// ---------------------------------------------------------------------------
__device__ __forceinline__ float softplusf(float x) {
  return (x > 20.f) ? x : log1pf(__expf(x));
}

__global__ __launch_bounds__(128)
void evi_gate(const float* __restrict__ X, const float* __restrict__ Wg,
              const float* __restrict__ bg, float* __restrict__ lg, int Ttot) {
  const int lane = threadIdx.x & 31, wave = threadIdx.x >> 5;
  const int tok = blockIdx.x * 4 + wave;
  if (tok >= Ttot) return;
  const float* x = X + (size_t)tok * DM;
  float h0 = 0.f, h1 = 0.f, h2 = 0.f, h3 = 0.f;
  for (int i = lane; i < DM; i += 32) {
    const float xv = x[i];
    h0 += xv * Wg[i * 4 + 0]; h1 += xv * Wg[i * 4 + 1];
    h2 += xv * Wg[i * 4 + 2]; h3 += xv * Wg[i * 4 + 3];
  }
#pragma unroll
  for (int off = 16; off > 0; off >>= 1) {
    h0 += __shfl_xor(h0, off, 32); h1 += __shfl_xor(h1, off, 32);
    h2 += __shfl_xor(h2, off, 32); h3 += __shfl_xor(h3, off, 32);
  }
  if (lane == 0) {
    const float mu = h0 + bg[0];
    const float v  = softplusf(h1 + bg[1]) + 1e-6f;
    const float a  = softplusf(h2 + bg[2]) + 1.f + 1e-6f;
    const float be = softplusf(h3 + bg[3]) + 1e-6f;
    const float var_ep = be / (v * (a - 1.f));
    const float g = (1.f / (1.f + __expf(-mu))) * __expf(-2.f * var_ep);
    lg[tok] = logf(fmaxf(g, 1e-6f));
  }
}

// ---------------------------------------------------------------------------
// Residual + LayerNorm: Y = LN(X + Dd) * g + b   (Dd nullable)
// ---------------------------------------------------------------------------
__global__ __launch_bounds__(256)
void res_ln(const float* __restrict__ X, const float* __restrict__ Dd,
            const float* __restrict__ g, const float* __restrict__ bb,
            float* __restrict__ Y, int rows) {
  const int row = blockIdx.x;
  if (row >= rows) return;
  const int tid = threadIdx.x, lane = tid & 31, wid = tid >> 5;
  const size_t base = (size_t)row * DM;
  float v0 = X[base + tid]       + (Dd ? Dd[base + tid] : 0.f);
  float v1 = X[base + tid + 256] + (Dd ? Dd[base + tid + 256] : 0.f);
  __shared__ float sm[8];
  float s = v0 + v1;
#pragma unroll
  for (int off = 16; off > 0; off >>= 1) s += __shfl_xor(s, off, 32);
  if (lane == 0) sm[wid] = s;
  __syncthreads();
  s = 0.f;
#pragma unroll
  for (int i = 0; i < 8; i++) s += sm[i];
  const float mean = s * (1.f / DM);
  __syncthreads();
  const float d0 = v0 - mean, d1 = v1 - mean;
  s = d0 * d0 + d1 * d1;
#pragma unroll
  for (int off = 16; off > 0; off >>= 1) s += __shfl_xor(s, off, 32);
  if (lane == 0) sm[wid] = s;
  __syncthreads();
  s = 0.f;
#pragma unroll
  for (int i = 0; i < 8; i++) s += sm[i];
  const float rstd = rsqrtf(s * (1.f / DM) + 1e-5f);
  Y[base + tid]       = d0 * rstd * g[tid] + bb[tid];
  Y[base + tid + 256] = d1 * rstd * g[tid + 256] + bb[tid + 256];
}

// ---------------------------------------------------------------------------
__global__ __launch_bounds__(512)
void seq_mean(const float* __restrict__ X, float* __restrict__ out, int T) {
  const int b = blockIdx.x, c = threadIdx.x;
  float s = 0.f;
  for (int t = 0; t < T; t++) s += X[((size_t)b * T + t) * DM + c];
  out[(size_t)b * DM + c] = s / (float)T;
}

__global__ __launch_bounds__(256)
void add_bcast(float* __restrict__ Dm, const float* __restrict__ chem, int T,
               size_t total) {
  const size_t idx = (size_t)blockIdx.x * 256 + threadIdx.x;
  if (idx >= total) return;
  const int c = (int)(idx % DM);
  const int b = (int)(idx / ((size_t)T * DM));
  Dm[idx] += chem[(size_t)b * DM + c];
}

__global__ __launch_bounds__(256)
void mutan_z(const float* __restrict__ bx, const float* __restrict__ by,
             float* __restrict__ z, int total) {
  const int i = blockIdx.x * 256 + threadIdx.x;
  if (i >= total) return;
  const int b = i / MUT, c = i % MUT;
  float s = 0.f;
#pragma unroll
  for (int r = 0; r < RANKM; r++) {
    const size_t o = (size_t)b * (RANKM * MUT) + r * MUT + c;
    s += tanhf(bx[o]) * tanhf(by[o]);
  }
  z[i] = s;
}

__global__ __launch_bounds__(32)
void head_out(const float* __restrict__ hf, const float* __restrict__ w2,
              const float* __restrict__ b2, float* __restrict__ out, int Bc) {
  const int b = blockIdx.x;
  if (b >= Bc) return;
  const int lane = threadIdx.x;
  float s = 0.f;
  for (int i = lane; i < DM; i += 32) s += hf[(size_t)b * DM + i] * w2[i];
#pragma unroll
  for (int off = 16; off > 0; off >>= 1) s += __shfl_xor(s, off, 32);
  if (lane == 0) out[b] = s + b2[0];
}

// ---------------------------------------------------------------------------
// Host orchestration
// ---------------------------------------------------------------------------
struct LayerP {
  const float *qx,*kx,*vx,*qy,*ky,*vy,*out_x,*out_y;
  const float *gx_w,*gx_b,*gy_w,*gy_b;
  const float *lnx_g,*lnx_b,*lny_g,*lny_b;
  const float *ffx_w,*ffx_b,*ffy_w,*ffy_b;
};
struct ModelP {
  const float *proj_p,*proj_d,*proj_c;
  LayerP L[2];
  const float *mutan_px,*mutan_py,*mutan_ln_g,*mutan_ln_b;
  const float *head_ln_g,*head_ln_b,*head_w1,*head_b1,*head_w2,*head_b2;
  const float *v_prot,*v_mol,*v_chem;
};

extern "C" void kernel_launch(void* const* d_in, const int* in_sizes, int n_in,
                              void* d_out, int out_size, void* d_ws,
                              size_t ws_size, hipStream_t stream) {
  (void)n_in; (void)out_size; (void)ws_size;
  const int Bc = 16, Mp = 2048, Nm = 256;

  auto F = [&](int i) { return (const float*)d_in[i]; };
  ModelP P;
  if (in_sizes[0] == 16 * 2048 * 1280) {
    // dict insertion order flattening
    P.v_prot = F(0); P.v_mol = F(2); P.v_chem = F(4);
    P.proj_p = F(5); P.proj_d = F(6); P.proj_c = F(7);
    for (int l = 0; l < 2; l++) {
      const int o = 8 + l * 20;
      LayerP& L = P.L[l];
      L.qx = F(o+0);  L.kx = F(o+1);  L.vx = F(o+2);
      L.qy = F(o+3);  L.ky = F(o+4);  L.vy = F(o+5);
      L.out_x = F(o+6); L.out_y = F(o+7);
      L.gx_w = F(o+8); L.gx_b = F(o+9); L.gy_w = F(o+10); L.gy_b = F(o+11);
      L.lnx_g = F(o+12); L.lnx_b = F(o+13); L.lny_g = F(o+14); L.lny_b = F(o+15);
      L.ffx_w = F(o+16); L.ffx_b = F(o+17); L.ffy_w = F(o+18); L.ffy_b = F(o+19);
    }
    P.mutan_px = F(48); P.mutan_py = F(49);
    P.mutan_ln_g = F(50); P.mutan_ln_b = F(51);
    P.head_ln_g = F(52); P.head_ln_b = F(53);
    P.head_w1 = F(54); P.head_b1 = F(55); P.head_w2 = F(56); P.head_b2 = F(57);
  } else {
    // jax pytree flattening (sorted dict keys at every level)
    P.head_b1 = F(2); P.head_b2 = F(3);
    P.head_ln_b = F(4); P.head_ln_g = F(5);
    P.head_w1 = F(6); P.head_w2 = F(7);
    for (int l = 0; l < 2; l++) {
      const int o = 8 + l * 20;
      LayerP& L = P.L[l];
      L.ffx_b = F(o+0); L.ffx_w = F(o+1); L.ffy_b = F(o+2); L.ffy_w = F(o+3);
      L.gx_b = F(o+4);  L.gx_w = F(o+5);  L.gy_b = F(o+6);  L.gy_w = F(o+7);
      L.kx = F(o+8);    L.ky = F(o+9);
      L.lnx_b = F(o+10); L.lnx_g = F(o+11); L.lny_b = F(o+12); L.lny_g = F(o+13);
      L.out_x = F(o+14); L.out_y = F(o+15);
      L.qx = F(o+16);   L.qy = F(o+17);   L.vx = F(o+18);   L.vy = F(o+19);
    }
    P.mutan_ln_b = F(48); P.mutan_ln_g = F(49);
    P.mutan_px = F(50); P.mutan_py = F(51);
    P.proj_c = F(52); P.proj_d = F(53); P.proj_p = F(54);
    P.v_chem = F(55); P.v_mol = F(56); P.v_prot = F(57);
  }

  // ---- workspace carve
  char* wsp = (char*)d_ws;
  size_t off = 0;
  auto alloc = [&](size_t bytes) -> void* {
    void* p = wsp + off;
    off = (off + bytes + 255) & ~(size_t)255;
    return p;
  };
  float* P0   = (float*)alloc((size_t)Bc * Mp * DM * 4);
  float* P1   = (float*)alloc((size_t)Bc * Mp * DM * 4);
  float* D0   = (float*)alloc((size_t)Bc * Nm * DM * 4);
  float* D1   = (float*)alloc((size_t)Bc * Nm * DM * 4);
  float* chem = (float*)alloc((size_t)Bc * DM * 4);
  uint16_t* Qx  = (uint16_t*)alloc((size_t)Bc * Nm * DM * 2);
  uint16_t* Kx  = (uint16_t*)alloc((size_t)Bc * Nm * DM * 2);
  uint16_t* VxT = (uint16_t*)alloc((size_t)Bc * Nm * DM * 2);
  uint16_t* Qy  = (uint16_t*)alloc((size_t)Bc * Mp * DM * 2);
  uint16_t* Ky  = (uint16_t*)alloc((size_t)Bc * Mp * DM * 2);
  uint16_t* VyT = (uint16_t*)alloc((size_t)Bc * Mp * DM * 2);
  float* AOx = (float*)alloc((size_t)Bc * Nm * DM * 4);
  float* AOy = (float*)alloc((size_t)Bc * Mp * DM * 4);
  float* Tx  = (float*)alloc((size_t)Bc * Nm * DM * 4);
  float* Ty  = (float*)alloc((size_t)Bc * Mp * DM * 4);
  float* lgx = (float*)alloc((size_t)Bc * Nm * 4);
  float* lgy = (float*)alloc((size_t)Bc * Mp * 4);
  float* hd  = (float*)alloc((size_t)Bc * DM * 4);
  float* hp  = (float*)alloc((size_t)Bc * DM * 4);
  float* bx  = (float*)alloc((size_t)Bc * RANKM * MUT * 4);
  float* by  = (float*)alloc((size_t)Bc * RANKM * MUT * 4);
  float* zb  = (float*)alloc((size_t)Bc * MUT * 4);
  float* z2  = (float*)alloc((size_t)Bc * MUT * 4);
  float* z3  = (float*)alloc((size_t)Bc * MUT * 4);
  float* hf  = (float*)alloc((size_t)Bc * DM * 4);

  auto gemm = [&](const float* A, const float* W, const float* bias, void* C,
                  int M, int N, int K, int act, int outmode, int T) {
    dim3 grid((N + 63) / 64, (M + 63) / 64);
    if (act == 0 && outmode == 0)
      gemm_wmma<0, 0><<<grid, 128, 0, stream>>>(A, W, bias, C, M, N, K, T);
    else if (act == 0 && outmode == 1)
      gemm_wmma<0, 1><<<grid, 128, 0, stream>>>(A, W, bias, C, M, N, K, T);
    else if (act == 0 && outmode == 2)
      gemm_wmma<0, 2><<<grid, 128, 0, stream>>>(A, W, bias, C, M, N, K, T);
    else
      gemm_wmma<1, 0><<<grid, 128, 0, stream>>>(A, W, bias, C, M, N, K, T);
  };

  // ---- input projections
  gemm(P.v_prot, P.proj_p, nullptr, P0, Bc * Mp, DM, 1280, 0, 0, 0);
  gemm(P.v_mol,  P.proj_d, nullptr, D0, Bc * Nm, DM, 300,  0, 0, 0);
  gemm(P.v_chem, P.proj_c, nullptr, chem, Bc, DM, 384, 0, 0, 0);
  {
    const size_t tot = (size_t)Bc * Nm * DM;
    add_bcast<<<(unsigned)((tot + 255) / 256), 256, 0, stream>>>(D0, chem, Nm, tot);
  }

  const float scale = 0.08838834764831845f;  // 1/sqrt(128)

  for (int l = 0; l < 2; l++) {
    const LayerP& L = P.L[l];
    evi_gate<<<(Bc * Nm + 3) / 4, 128, 0, stream>>>(D0, L.gx_w, L.gx_b, lgx, Bc * Nm);
    evi_gate<<<(Bc * Mp + 3) / 4, 128, 0, stream>>>(P0, L.gy_w, L.gy_b, lgy, Bc * Mp);

    gemm(D0, L.qx, nullptr, Qx,  Bc * Nm, DM, DM, 0, 1, 0);
    gemm(D0, L.kx, nullptr, Kx,  Bc * Nm, DM, DM, 0, 1, 0);
    gemm(D0, L.vx, nullptr, VxT, Bc * Nm, DM, DM, 0, 2, Nm);
    gemm(P0, L.qy, nullptr, Qy,  Bc * Mp, DM, DM, 0, 1, 0);
    gemm(P0, L.ky, nullptr, Ky,  Bc * Mp, DM, DM, 0, 1, 0);
    gemm(P0, L.vy, nullptr, VyT, Bc * Mp, DM, DM, 0, 2, Mp);

    const int t1 = Bc * NH * (Nm / 16);
    attn_wmma<<<(t1 + 3) / 4, 128, 0, stream>>>(Qx, Ky, VyT, lgy, AOx, Bc, Nm, Mp, scale);
    const int t2 = Bc * NH * (Mp / 16);
    attn_wmma<<<(t2 + 3) / 4, 128, 0, stream>>>(Qy, Kx, VxT, lgx, AOy, Bc, Mp, Nm, scale);

    // x (mol) side
    gemm(AOx, L.out_x, nullptr, Tx, Bc * Nm, DM, DM, 0, 0, 0);
    res_ln<<<Bc * Nm, 256, 0, stream>>>(D0, Tx, L.lnx_g, L.lnx_b, D1, Bc * Nm);
    gemm(D1, L.ffx_w, L.ffx_b, Tx, Bc * Nm, DM, DM, 1, 0, 0);
    res_ln<<<Bc * Nm, 256, 0, stream>>>(D1, Tx, L.lnx_g, L.lnx_b, D0, Bc * Nm);
    // y (prot) side
    gemm(AOy, L.out_y, nullptr, Ty, Bc * Mp, DM, DM, 0, 0, 0);
    res_ln<<<Bc * Mp, 256, 0, stream>>>(P0, Ty, L.lny_g, L.lny_b, P1, Bc * Mp);
    gemm(P1, L.ffy_w, L.ffy_b, Ty, Bc * Mp, DM, DM, 1, 0, 0);
    res_ln<<<Bc * Mp, 256, 0, stream>>>(P1, Ty, L.lny_g, L.lny_b, P0, Bc * Mp);
  }

  // ---- pooling + MUTAN + head
  seq_mean<<<Bc, 512, 0, stream>>>(D0, hd, Nm);
  seq_mean<<<Bc, 512, 0, stream>>>(P0, hp, Mp);
  gemm(hd, P.mutan_px, nullptr, bx, Bc, RANKM * MUT, DM, 0, 0, 0);
  gemm(hp, P.mutan_py, nullptr, by, Bc, RANKM * MUT, DM, 0, 0, 0);
  mutan_z<<<(Bc * MUT + 255) / 256, 256, 0, stream>>>(bx, by, zb, Bc * MUT);
  res_ln<<<Bc, 256, 0, stream>>>(zb, nullptr, P.mutan_ln_g, P.mutan_ln_b, z2, Bc);
  res_ln<<<Bc, 256, 0, stream>>>(z2, nullptr, P.head_ln_g, P.head_ln_b, z3, Bc);
  gemm(z3, P.head_w1, P.head_b1, hf, Bc, DM, DM, 1, 0, 0);
  head_out<<<Bc, 32, 0, stream>>>(hf, P.head_w2, P.head_b2, (float*)d_out, Bc);
}
